// DUQHead_44444321579350
// MI455X (gfx1250) — matrix operand order repaired and checked by reading, not compiled, not run on hardware
//
#include <hip/hip_runtime.h>
#include <hip/hip_bf16.h>

typedef __attribute__((ext_vector_type(16))) __bf16 v16bf;
typedef __attribute__((ext_vector_type(8)))  float  v8f;

union BF16x16 { v16bf v; unsigned u[8]; };

__device__ __forceinline__ unsigned short f2bf(float f) {
    unsigned u = __float_as_uint(f);
    u += 0x7FFFu + ((u >> 16) & 1u);   // round-to-nearest-even
    return (unsigned short)(u >> 16);
}

// --- CDNA5 async Global->LDS (ASYNCcnt-tracked), via inline asm -------------
__device__ __forceinline__ void async_load_b128(const void* lds_ptr, const void* gptr) {
    unsigned loff = (unsigned)(size_t)lds_ptr;
    unsigned long long ga = (unsigned long long)gptr;
    asm volatile("global_load_async_to_lds_b128 %0, %1, off"
                 :: "v"(loff), "v"(ga) : "memory");
}
#define WAIT_ASYNC_2() asm volatile("s_wait_asynccnt 0x2" ::: "memory")
#define WAIT_ASYNC_0() asm volatile("s_wait_asynccnt 0x0" ::: "memory")

__device__ __forceinline__ v8f wmma_bf16(const BF16x16& a, const BF16x16& b, v8f c) {
    return __builtin_amdgcn_wmma_f32_16x16x32_bf16(
        false, a.v, false, b.v, (short)0, c, false, false);
}

// ---------------------------------------------------------------------------
// Pack f32 [K x N] row-major -> dword [K/2 x N], dword = {bf16(k), bf16(k+1)}
// ---------------------------------------------------------------------------
__global__ void pack_bf16_pairs_kernel(const float* __restrict__ src,
                                       unsigned* __restrict__ dst,
                                       int N, long long total) {
    long long i = (long long)blockIdx.x * blockDim.x + threadIdx.x;
    if (i >= total) return;
    long long kk = i / N, n = i % N;
    float a = src[(2 * kk) * (long long)N + n];
    float b = src[(2 * kk + 1) * (long long)N + n];
    dst[i] = (unsigned)f2bf(a) | ((unsigned)f2bf(b) << 16);
}

// ---------------------------------------------------------------------------
// LayerNorm (f32 in) -> bf16 out, one block per row
// ---------------------------------------------------------------------------
template<int COLS, int PER>
__global__ __launch_bounds__(256) void ln_to_bf16_kernel(
    const float* __restrict__ x, const float* __restrict__ g,
    const float* __restrict__ beta, unsigned short* __restrict__ y) {
    __shared__ float red[256];
    int row = blockIdx.x;
    const float* xr = x + (size_t)row * COLS;
    float v[PER];
    float s = 0.f, s2 = 0.f;
#pragma unroll
    for (int i = 0; i < PER; ++i) {
        int idx = threadIdx.x + i * 256;
        v[i] = xr[idx];
        s += v[i]; s2 += v[i] * v[i];
    }
    red[threadIdx.x] = s; __syncthreads();
    for (int off = 128; off > 0; off >>= 1) {
        if (threadIdx.x < off) red[threadIdx.x] += red[threadIdx.x + off];
        __syncthreads();
    }
    float mu = red[0] * (1.0f / COLS);
    __syncthreads();
    red[threadIdx.x] = s2; __syncthreads();
    for (int off = 128; off > 0; off >>= 1) {
        if (threadIdx.x < off) red[threadIdx.x] += red[threadIdx.x + off];
        __syncthreads();
    }
    float var = red[0] * (1.0f / COLS) - mu * mu;
    float rs = rsqrtf(var + 1e-5f);
#pragma unroll
    for (int i = 0; i < PER; ++i) {
        int idx = threadIdx.x + i * 256;
        float o = (v[i] - mu) * rs * g[idx] + beta[idx];
        y[(size_t)row * COLS + idx] = f2bf(o);
    }
}

// ---------------------------------------------------------------------------
// Fragment helpers
// ---------------------------------------------------------------------------
__device__ __forceinline__ void load_a_frag(BF16x16& Af, const unsigned short* arow,
                                            int kk, int hi) {
    int ka = kk + (hi << 3);
#pragma unroll
    for (int j = 0; j < 8; ++j) {
        int k0 = ka + ((j >> 2) << 4) + ((j & 3) << 1);
        Af.u[j] = *(const unsigned*)(arow + k0);
    }
}

__device__ __forceinline__ void load_b_frag(BF16x16& Bf, const unsigned* ldsB,
                                            int hi, int colL) {
#pragma unroll
    for (int j = 0; j < 8; ++j)
        Bf.u[j] = ldsB[(((hi << 2) + j + ((j >> 2) << 2)) << 6) + colL];
}

// Stage a 32-packed-row x 64-dword B chunk (8 KB): two async b128 per thread.
__device__ __forceinline__ void stage_b_chunk(const unsigned* __restrict__ Bpk, int ldN,
                                              int kb0, int n0, unsigned* buf) {
#pragma unroll
    for (int i = 0; i < 2; ++i) {
        int gr = threadIdx.x + (i << 8);   // 16B granule 0..511
        int r  = gr >> 4;                  // packed row 0..31
        int cg = (gr & 15) << 2;           // dword col group
        async_load_b128(buf + (r << 6) + cg,
                        Bpk + (size_t)(kb0 + r) * ldN + n0 + cg);
    }
}

// ---------------------------------------------------------------------------
// bf16 WMMA GEMM, double-buffered async B staging.
// 8 waves/block, wave grid 2(M) x 4(N); block tile 64 x 64; K-chunk 64.
// Each wave: two 16x16 accumulators (rows m, m+32) -> 4 WMMAs per barrier.
// ---------------------------------------------------------------------------
template<bool GELU, bool OUT_BF16>
__global__ __launch_bounds__(256) void gemm_bf16_kernel(
    const unsigned short* __restrict__ A, const unsigned* __restrict__ Bpk,
    const float* __restrict__ bias, void* __restrict__ out,
    int M, int N, int K) {
    __shared__ __align__(16) unsigned ldsB[2][32 * 64];

    int lane = threadIdx.x & 31;
    int w    = threadIdx.x >> 5;
    int wm = w >> 2, wn = w & 3;
    int mbase = blockIdx.y * 64 + wm * 16;
    int n0wg  = blockIdx.x * 64;
    int ml = lane & 15, hi = lane >> 4;
    const unsigned short* arow0 = A + (size_t)(mbase + ml) * K;
    const unsigned short* arow1 = A + (size_t)(mbase + 32 + ml) * K;
    int colL = wn * 16 + ml;

    int nchunks = K >> 6;
    stage_b_chunk(Bpk, N, 0, n0wg, ldsB[0]);

    v8f acc0 = {}, acc1 = {};
    for (int it = 0; it < nchunks; ++it) {
        if (it + 1 < nchunks) {
            stage_b_chunk(Bpk, N, (it + 1) << 5, n0wg, ldsB[(it + 1) & 1]);
            WAIT_ASYNC_2();
        } else {
            WAIT_ASYNC_0();
        }
        __syncthreads();
        const unsigned* buf = ldsB[it & 1];
#pragma unroll
        for (int ks = 0; ks < 2; ++ks) {
            int kk = (it << 6) + (ks << 5);
            BF16x16 Af0, Af1, Bf;
            load_a_frag(Af0, arow0, kk, hi);
            load_a_frag(Af1, arow1, kk, hi);
            load_b_frag(Bf, buf + (ks << 10), hi, colL);
            acc0 = wmma_bf16(Af0, Bf, acc0);
            acc1 = wmma_bf16(Af1, Bf, acc1);
        }
        __syncthreads();
    }
#pragma unroll
    for (int r = 0; r < 8; ++r) {
        int n = n0wg + colL;
        int m = mbase + (hi << 3) + r;
        float v0 = acc0[r] + bias[n];
        float v1 = acc1[r] + bias[n];
        if (GELU) {
            v0 = 0.5f * v0 * (1.f + erff(v0 * 0.70710678118654752f));
            v1 = 0.5f * v1 * (1.f + erff(v1 * 0.70710678118654752f));
        }
        if (OUT_BF16) {
            ((unsigned short*)out)[(size_t)m * N + n]        = f2bf(v0);
            ((unsigned short*)out)[(size_t)(m + 32) * N + n] = f2bf(v1);
        } else {
            ((float*)out)[(size_t)m * N + n]        = v0;
            ((float*)out)[(size_t)(m + 32) * N + n] = v1;
        }
    }
}

// ---------------------------------------------------------------------------
// Per-class transform + RBF distance (dominant 26.8 GFLOP):
//   weighted[b,c,:] = z[b,:] @ class_weight[c];  lk[b,c] = -50*||w - cent_c||^2
// grid = (B/64, C). 16 pipelined iterations over (ngroup g:4, kchunk:4),
// fused squared-diff epilogue after each 64-col group.
// ---------------------------------------------------------------------------
__global__ __launch_bounds__(256) void duq_dist_kernel(
    const unsigned short* __restrict__ z, const unsigned* __restrict__ cwpk,
    const float* __restrict__ cent, float* __restrict__ lk) {
    const int E = 256, C = 200;
    __shared__ __align__(16) unsigned ldsB[2][32 * 64];
    __shared__ float part[64][4];

    int lane = threadIdx.x & 31;
    int w    = threadIdx.x >> 5;
    int wm = w >> 2, wn = w & 3;
    int c  = blockIdx.y;
    int mbase = blockIdx.x * 64 + wm * 16;
    int ml = lane & 15, hi = lane >> 4;
    const unsigned short* arow0 = z + (size_t)(mbase + ml) * E;
    const unsigned short* arow1 = z + (size_t)(mbase + 32 + ml) * E;
    const unsigned* Bc = cwpk + (size_t)c * (E / 2) * E;
    const float* cc = cent + (size_t)c * E;
    int colL = wn * 16 + ml;

    // it = g*4 + kc : g = 64-col N-group, kc = 64-wide K-chunk
    stage_b_chunk(Bc, E, 0, 0, ldsB[0]);

    float sq0[8] = {0.f,0.f,0.f,0.f,0.f,0.f,0.f,0.f};
    float sq1[8] = {0.f,0.f,0.f,0.f,0.f,0.f,0.f,0.f};
    v8f acc0 = {}, acc1 = {};
    for (int it = 0; it < 16; ++it) {
        if (it + 1 < 16) {
            int nt = it + 1;
            stage_b_chunk(Bc, E, (nt & 3) << 5, (nt >> 2) << 6, ldsB[nt & 1]);
            WAIT_ASYNC_2();
        } else {
            WAIT_ASYNC_0();
        }
        __syncthreads();
        const unsigned* buf = ldsB[it & 1];
#pragma unroll
        for (int ks = 0; ks < 2; ++ks) {
            int kk = ((it & 3) << 6) + (ks << 5);
            BF16x16 Af0, Af1, Bf;
            load_a_frag(Af0, arow0, kk, hi);
            load_a_frag(Af1, arow1, kk, hi);
            load_b_frag(Bf, buf + (ks << 10), hi, colL);
            acc0 = wmma_bf16(Af0, Bf, acc0);
            acc1 = wmma_bf16(Af1, Bf, acc1);
        }
        __syncthreads();
        if ((it & 3) == 3) {                 // finished one 64-col group
            int n = ((it >> 2) << 6) + colL;
            float cv = cc[n];
#pragma unroll
            for (int r = 0; r < 8; ++r) {
                float d0 = acc0[r] - cv;
                float d1 = acc1[r] - cv;
                sq0[r] += d0 * d0;
                sq1[r] += d1 * d1;
            }
            acc0 = {};
            acc1 = {};
        }
    }
    // reduce squared sums across the 16 lanes of each half (same M rows)
#pragma unroll
    for (int r = 0; r < 8; ++r)
        for (int off = 1; off < 16; off <<= 1) {
            sq0[r] += __shfl_xor(sq0[r], off, 32);
            sq1[r] += __shfl_xor(sq1[r], off, 32);
        }

    if (lane == 0) {
#pragma unroll
        for (int r = 0; r < 8; ++r) {
            part[wm * 16 + r][wn]      = sq0[r];
            part[32 + wm * 16 + r][wn] = sq1[r];
        }
    }
    if (lane == 16) {
#pragma unroll
        for (int r = 0; r < 8; ++r) {
            part[wm * 16 + 8 + r][wn]      = sq0[r];
            part[32 + wm * 16 + 8 + r][wn] = sq1[r];
        }
    }
    __syncthreads();
    if (threadIdx.x < 64) {
        int m = threadIdx.x;
        float s = part[m][0] + part[m][1] + part[m][2] + part[m][3];
        int b = blockIdx.x * 64 + m;
        lk[(size_t)b * C + c] = -50.f * s;   // -dist/(2*0.1^2)
    }
}

// ---------------------------------------------------------------------------
// softmax / exp / uncertainty; one block per batch row
// out: [probs BC][probs BC][kernel_vals BC][uncertainty B]
// ---------------------------------------------------------------------------
__global__ __launch_bounds__(256) void duq_softmax_kernel(
    const float* __restrict__ lk, float* __restrict__ out, int B, int C) {
    __shared__ float red[256];
    int b = blockIdx.x;
    int c = threadIdx.x;
    float v = (c < C) ? lk[(size_t)b * C + c] : -3.402823e38f;
    red[c] = v; __syncthreads();
    for (int off = 128; off > 0; off >>= 1) {
        if (c < off) red[c] = fmaxf(red[c], red[c + off]);
        __syncthreads();
    }
    float mx = red[0]; __syncthreads();
    float e = (c < C) ? expf(v - mx) : 0.f;
    red[c] = e; __syncthreads();
    for (int off = 128; off > 0; off >>= 1) {
        if (c < off) red[c] += red[c + off];
        __syncthreads();
    }
    float s = red[0];
    size_t BC = (size_t)B * C;
    if (c < C) {
        float p  = e / s;
        float kv = expf(v);
        out[(size_t)b * C + c]            = p;
        out[BC + (size_t)b * C + c]       = p;
        out[2 * BC + (size_t)b * C + c]   = kv;
    }
    if (c == 0) out[3 * BC + b] = 1.f - expf(mx);
}

// ---------------------------------------------------------------------------
extern "C" void kernel_launch(void* const* d_in, const int* in_sizes, int n_in,
                              void* d_out, int out_size, void* d_ws, size_t ws_size,
                              hipStream_t stream) {
    (void)in_sizes; (void)n_in; (void)out_size; (void)ws_size;
    const int B = 1024, EMBED = 768, HIDDEN = 2048, C = 200, E = 256;

    const float* features = (const float*)d_in[0];
    const float* ln1_g    = (const float*)d_in[1];
    const float* ln1_b    = (const float*)d_in[2];
    const float* W1       = (const float*)d_in[3];
    const float* b1       = (const float*)d_in[4];
    const float* ln2_g    = (const float*)d_in[5];
    const float* ln2_b    = (const float*)d_in[6];
    const float* W2       = (const float*)d_in[7];
    const float* b2       = (const float*)d_in[8];
    const float* W3       = (const float*)d_in[9];
    const float* b3       = (const float*)d_in[10];
    const float* cw       = (const float*)d_in[11];
    const float* cent     = (const float*)d_in[12];

    char* ws = (char*)d_ws;
    size_t off = 0;
    auto alloc = [&](size_t bytes) -> void* {
        void* p = ws + off;
        off += (bytes + 255) & ~(size_t)255;
        return p;
    };
    unsigned*       w1pk = (unsigned*)      alloc((size_t)(EMBED / 2) * HIDDEN * 4);
    unsigned*       w2pk = (unsigned*)      alloc((size_t)(HIDDEN / 2) * HIDDEN * 4);
    unsigned*       w3pk = (unsigned*)      alloc((size_t)(HIDDEN / 2) * E * 4);
    unsigned*       cwpk = (unsigned*)      alloc((size_t)C * (E / 2) * E * 4);
    unsigned short* xln  = (unsigned short*)alloc((size_t)B * EMBED * 2);
    float*          h1   = (float*)         alloc((size_t)B * HIDDEN * 4);
    unsigned short* h1ln = (unsigned short*)alloc((size_t)B * HIDDEN * 2);
    unsigned short* h2   = (unsigned short*)alloc((size_t)B * HIDDEN * 2);
    unsigned short* zbf  = (unsigned short*)alloc((size_t)B * E * 2);
    float*          lkbf = (float*)         alloc((size_t)B * C * 4);

    // --- weight packing (f32 -> packed bf16 pairs) ---
    {
        long long t1 = (long long)(EMBED / 2) * HIDDEN;
        pack_bf16_pairs_kernel<<<(unsigned)((t1 + 255) / 256), 256, 0, stream>>>(W1, w1pk, HIDDEN, t1);
        long long t2 = (long long)(HIDDEN / 2) * HIDDEN;
        pack_bf16_pairs_kernel<<<(unsigned)((t2 + 255) / 256), 256, 0, stream>>>(W2, w2pk, HIDDEN, t2);
        long long t3 = (long long)(HIDDEN / 2) * E;
        pack_bf16_pairs_kernel<<<(unsigned)((t3 + 255) / 256), 256, 0, stream>>>(W3, w3pk, E, t3);
        long long t4 = (long long)C * (E / 2) * E;
        pack_bf16_pairs_kernel<<<(unsigned)((t4 + 255) / 256), 256, 0, stream>>>(cw, cwpk, E, t4);
    }

    // --- MLP ---
    ln_to_bf16_kernel<768, 3><<<B, 256, 0, stream>>>(features, ln1_g, ln1_b, xln);

    gemm_bf16_kernel<true, false><<<dim3(HIDDEN / 64, B / 64), 256, 0, stream>>>(
        xln, w1pk, b1, h1, B, HIDDEN, EMBED);

    ln_to_bf16_kernel<2048, 8><<<B, 256, 0, stream>>>(h1, ln2_g, ln2_b, h1ln);

    gemm_bf16_kernel<true, true><<<dim3(HIDDEN / 64, B / 64), 256, 0, stream>>>(
        h1ln, w2pk, b2, h2, B, HIDDEN, HIDDEN);

    gemm_bf16_kernel<false, true><<<dim3(E / 64, B / 64), 256, 0, stream>>>(
        h2, w3pk, b3, zbf, B, E, HIDDEN);

    // --- per-class transform + RBF distances ---
    duq_dist_kernel<<<dim3(B / 64, C), 256, 0, stream>>>(zbf, cwpk, cent, lkbf);

    // --- softmax / outputs ---
    duq_softmax_kernel<<<B, 256, 0, stream>>>(lkbf, (float*)d_out, B, C);
}
